// GlobalPointer_7181185318930
// MI455X (gfx1250) — matrix operand order repaired and checked by reading, not compiled, not run on hardware
//
#include <hip/hip_runtime.h>
#include <hip/hip_bf16.h>

// ---- problem constants (from reference) ----
#define B_    16
#define S_    512
#define DIN   768
#define HID_  64
#define OUT_  12
#define N1    1536          // HID*OUT*2
#define M_    (B_ * S_)     // 8192
#define NEGC  1000000000000.0f
#define NTRIG (S_ * 32)     // 16384 (cos,sin) pairs

typedef __attribute__((ext_vector_type(16))) _Float16 v16h;
typedef __attribute__((ext_vector_type(8)))  _Float16 v8h;
typedef __attribute__((ext_vector_type(8)))  float    v8f;

union FragAB { v16h v; v8h h[2]; };

struct CS { float c, s; };   // 8-byte (cos,sin) pair -> one b64 load

// A (16x32 f16) fragment: lane m = lane&15; halves K = hi*16+0..15
__device__ __forceinline__ void loadA(FragAB& f, const _Float16* arow, int kb) {
  f.h[0] = *(const v8h*)(arow + kb);
  f.h[1] = *(const v8h*)(arow + kb + 8);
}
// B (32x16 f16) fragment from row-major [n][k]: lane n = lane&15; halves K = hi*16+0..15
__device__ __forceinline__ void loadB(FragAB& f, const _Float16* bcol, int kb) {
  f.h[0] = *(const v8h*)(bcol + kb);
  f.h[1] = *(const v8h*)(bcol + kb + 8);
}

// ---------------------------------------------------------------------------
// prep: x f32->f16; W f32->f16 transposed to Wt[n][k]; RoPE trig LUT.
// ---------------------------------------------------------------------------
__global__ void prep_kernel(const float* __restrict__ x, const float* __restrict__ W,
                            _Float16* __restrict__ xh, _Float16* __restrict__ Wt,
                            CS* __restrict__ trig) {
  size_t i = (size_t)blockIdx.x * blockDim.x + threadIdx.x;
  const size_t nx = (size_t)M_ * DIN;
  const size_t nw = (size_t)N1 * DIN;
  if (i < nx) {
    xh[i] = (_Float16)__builtin_nontemporal_load(x + i);
  } else if (i < nx + nw) {
    size_t j = i - nx;                 // Wt index: [n][k]
    int n = (int)(j / DIN);
    int k = (int)(j % DIN);
    Wt[j] = (_Float16)W[(size_t)k * N1 + n];
  } else if (i < nx + nw + NTRIG) {
    int t = (int)(i - nx - nw);
    int s = t >> 5;                    // position 0..511
    int j = t & 31;                    // frequency index
    float invf = exp2f(-(float)j * 0.41524101186092029f);  // 10000^(-2j/64)
    float ang = (float)s * invf;
    CS cs; cs.c = cosf(ang); cs.s = sinf(ang);
    trig[t] = cs;
  }
}

// ---------------------------------------------------------------------------
// Stage 1: proj = x@W + b, then RoPE; write q/k as f16 in [2][B][OUT][S][HID].
// Wave computes 16(M) x 64(N); K-loop of 24 v_wmma_f32_16x16x32_f16.
// ---------------------------------------------------------------------------
__global__ void __launch_bounds__(128)
proj_rope_kernel(const _Float16* __restrict__ xh, const _Float16* __restrict__ Wt,
                 const float* __restrict__ bias, const CS* __restrict__ trig,
                 _Float16* __restrict__ qk) {
  const int wave = (blockIdx.x << 2) + (threadIdx.x >> 5);
  const int lane = threadIdx.x & 31;
  const int lo   = lane & 15;
  const int hi   = lane >> 4;

  const int nStrip = wave % 24;            // 24 strips of 64 cols (N1=1536)
  const int mTile  = wave / 24;            // 512 tiles of 16 rows (M=8192)
  const int tileN  = nStrip * 64;
  const int tileM  = mTile * 16;

  const _Float16* arow = xh + (size_t)(tileM + lo) * DIN + hi * 16;
  const _Float16* bcol[4];
#pragma unroll
  for (int g = 0; g < 4; ++g)
    bcol[g] = Wt + (size_t)(tileN + g * 16 + lo) * DIN + hi * 16;

  v8f acc[4] = {};
#pragma unroll
  for (int kb = 0; kb < DIN; kb += 32) {
    FragAB a;
    loadA(a, arow, kb);
#pragma unroll
    for (int g = 0; g < 4; ++g) {
      FragAB bf;
      loadB(bf, bcol[g], kb);
      acc[g] = __builtin_amdgcn_wmma_f32_16x16x32_f16(
          false, a.v, false, bf.v, (short)0, acc[g], false, false);
    }
  }

  // Epilogue: bias + RoPE (table lookup), store f16 q/k.
#pragma unroll
  for (int g = 0; g < 4; ++g) {
    const int n   = tileN + g * 16 + lo;
    const int o   = n >> 7;            // head index (stride 128 in N)
    const int c   = n & 127;
    const int isK = (c >= 64) ? 1 : 0;
    const int d   = c & 63;
    const int j   = d & 31;            // cos/sin index = d % 32
    const float sgn = (d & 1) ? 1.0f : -1.0f;  // w2: even d: -w[d+1]; odd: +w[d-1]
    const float bb  = bias[n];
#pragma unroll
    for (int r = 0; r < 8; ++r) {
      const int m    = tileM + hi * 8 + r;
      const int bIdx = m >> 9;                 // m / 512
      const int s    = m & (S_ - 1);
      float w = acc[g][r] + bb;
      float p = __shfl_xor(w, 1, 32);          // partner channel (d^1), same row
      const CS cs = trig[(s << 5) + j];        // b64 from 128KB L2-resident table
      const float val = w * cs.c + sgn * p * cs.s;
      qk[(((((size_t)(isK * B_ + bIdx)) * OUT_ + o) * S_ + s) * HID_) + d] = (_Float16)val;
    }
  }
}

// ---------------------------------------------------------------------------
// Stage 2: per (b,h): logits = Q(512x64) @ K^T, pad-mask, tril(-1) mask,
// scale 1/8. Block = 4 waves sharing one 64-wide K-tile of one (b,h):
// the 8KB K-tile (64 rows x 64 k, f16) is staged in LDS once per block with
// global_load_async_to_lds_b128 (ASYNCcnt), then each wave computes a
// 16(M) x 64(N) tile with B-fragments as conflict-free ds_load_b128
// (row stride 144B = 36 banks). Output streamed with non-temporal stores.
// ---------------------------------------------------------------------------
#define KROW_PAD_H 72   // 64 halves + 8 pad halves = 144 B row stride

__global__ void __launch_bounds__(128)
logits_kernel(const _Float16* __restrict__ qk, const float* __restrict__ mask,
              float* __restrict__ out) {
  __shared__ _Float16 ktile[64 * KROW_PAD_H];   // 9216 B

  const int tid  = threadIdx.x;
  const int wid  = tid >> 5;
  const int lane = tid & 31;
  const int lo   = lane & 15;
  const int hi   = lane >> 4;

  // block -> (b, o, nStrip, mGroup); wave -> mTile = mGroup*4 + wid
  const int blk    = blockIdx.x;
  const int nStrip = blk & 7;                // 8 strips of 64 cols
  const int mGroup = (blk >> 3) & 7;         // 8 groups of 4 M-tiles
  const int bo     = blk >> 6;               // 0..191
  const int b      = bo / OUT_;
  const int o      = bo % OUT_;

  const _Float16* Q  = qk + (((size_t)b * OUT_ + o) * S_) * HID_;
  const _Float16* Km = qk + (((size_t)(B_ + b) * OUT_ + o) * S_) * HID_;

  const int tileM = (mGroup * 4 + wid) * 16;
  const int tileN = nStrip * 64;

  // ---- async-stage the shared K-tile: rows n = tileN..tileN+63, all 64 k ----
  // 512 16-byte chunks; 128 threads x 4 chunks. Generic __shared__ pointers
  // carry the LDS byte offset in their low 32 bits (flat LDS aperture).
  {
    const unsigned ldsBase = (unsigned)(size_t)(&ktile[0]);
#pragma unroll
    for (int i = 0; i < 4; ++i) {
      const int c    = tid + 128 * i;        // chunk id 0..511
      const int row  = c >> 3;               // 0..63
      const int off16 = c & 7;               // 16B chunk within row
      const unsigned ldsOff = ldsBase + (unsigned)(row * (KROW_PAD_H * 2) + off16 * 16);
      const unsigned long long ga =
          (unsigned long long)(const void*)(Km + (size_t)(tileN + row) * HID_ + off16 * 8);
      asm volatile("global_load_async_to_lds_b128 %0, %1, off"
                   :: "v"(ldsOff), "v"(ga) : "memory");
    }
    asm volatile("s_wait_asynccnt 0x0" ::: "memory");
    __syncthreads();
  }

  const _Float16* arow = Q + (size_t)(tileM + lo) * HID_ + hi * 16;

  v8f acc[4] = {};
#pragma unroll
  for (int kb = 0; kb < HID_; kb += 32) {
    FragAB a;
    loadA(a, arow, kb);
#pragma unroll
    for (int g = 0; g < 4; ++g) {
      // B fragment from LDS: lane row = g*16+lo, halves k = kb + hi*16 + 0..15
      const _Float16* p = &ktile[(g * 16 + lo) * KROW_PAD_H + kb + hi * 16];
      FragAB bf;
      bf.h[0] = *(const v8h*)(p);
      bf.h[1] = *(const v8h*)(p + 8);
      acc[g] = __builtin_amdgcn_wmma_f32_16x16x32_f16(
          false, a.v, false, bf.v, (short)0, acc[g], false, false);
    }
  }

  float* obase = out + (((size_t)b * OUT_ + o) * S_) * S_;
#pragma unroll
  for (int g = 0; g < 4; ++g) {
    const int n     = tileN + g * 16 + lo;
    const float pad = mask[(size_t)b * S_ + n];
    const float negm = (1.0f - pad) * NEGC;
#pragma unroll
    for (int r = 0; r < 8; ++r) {
      const int m = tileM + hi * 8 + r;
      float v = acc[g][r] * pad - negm;        // pad mask
      v = (m > n) ? (v - NEGC) : v;            // tril(ones,-1) mask
      __builtin_nontemporal_store(v * 0.125f, obase + (size_t)m * S_ + n);
    }
  }
}

// ---------------------------------------------------------------------------
extern "C" void kernel_launch(void* const* d_in, const int* in_sizes, int n_in,
                              void* d_out, int out_size, void* d_ws, size_t ws_size,
                              hipStream_t stream) {
  const float* x    = (const float*)d_in[0];   // (16,512,768) f32
  const float* mask = (const float*)d_in[1];   // (16,512)     f32
  const float* W    = (const float*)d_in[2];   // (768,1536)   f32
  const float* bias = (const float*)d_in[3];   // (1536,)      f32
  float* out        = (float*)d_out;           // (16,12,512,512) f32

  // workspace layout: xh (f16) | Wt (f16) | qk (f16) | trig (float2) ~40.1 MB
  _Float16* xh = (_Float16*)d_ws;
  _Float16* Wt = xh + (size_t)M_ * DIN;
  _Float16* qk = Wt + (size_t)N1 * DIN;                  // [2][B][OUT][S][HID]
  CS*       tr = (CS*)(qk + (size_t)2 * B_ * OUT_ * S_ * HID_);

  {
    const size_t tot = (size_t)M_ * DIN + (size_t)N1 * DIN + NTRIG;
    const int thr = 256;
    const int blk = (int)((tot + thr - 1) / thr);
    prep_kernel<<<blk, thr, 0, stream>>>(x, W, xh, Wt, tr);
  }
  {
    const int waves = 512 * 24;                // M-tiles * N-strips
    proj_rope_kernel<<<waves / 4, 128, 0, stream>>>(xh, Wt, bias, tr, qk);
  }
  {
    const int blocks = 192 * 8 * 8;            // (b,h) * mGroups * nStrips
    logits_kernel<<<blocks, 128, 0, stream>>>(qk, mask, out);
  }
}
